// VqQuantizer_29180007809309
// MI455X (gfx1250) — matrix operand order repaired and checked by reading, not compiled, not run on hardware
//
#include <hip/hip_runtime.h>
#include <hip/hip_bf16.h>

typedef __attribute__((ext_vector_type(16))) __bf16       v16bf;
typedef __attribute__((ext_vector_type(8)))  float        v8f;
typedef __attribute__((ext_vector_type(4)))  unsigned int v4u;
typedef __attribute__((ext_vector_type(8)))  int          v8i;
typedef __attribute__((ext_vector_type(4)))  int          v4i;

#define N_EMB     4096
#define EMB_DIM   256
#define HW        1024          // 32*32
#define NVEC      32768         // 32 * HW
#define ZQ_ELEMS  8388608       // 32*256*32*32
#define ROWS      32            // z rows per workgroup
#define KTILES    (N_EMB / 16)  // 256
#define CHUNKS    (EMB_DIM / 32)// 8

#if defined(__has_builtin)
#if __has_builtin(__builtin_amdgcn_tensor_load_to_lds) && \
    __has_builtin(__builtin_amdgcn_s_wait_tensorcnt)
#define USE_TDM 1
#endif
#endif
#ifndef USE_TDM
#define USE_TDM 0
#endif

// ---------------------------------------------------------------------------
// Kernel 1: per-code squared norms  ||e_k||^2   (one wave per code row)
// ---------------------------------------------------------------------------
__global__ void k_enorm(const float* __restrict__ E, float* __restrict__ enorm) {
    int k = blockIdx.x;
    int lane = threadIdx.x;                 // 32 threads
    const float* row = E + (size_t)k * EMB_DIM;
    float s = 0.f;
    for (int i = lane; i < EMB_DIM; i += 32) { float v = row[i]; s += v * v; }
    for (int off = 16; off; off >>= 1) s += __shfl_down(s, off);
    if (lane == 0) enorm[k] = s;
}

// ---------------------------------------------------------------------------
// Kernel 2: repack embedding into bf16 hi/lo images in WMMA B-operand layout.
// B 32x16 (Kdim x N), contiguous k (A matches): lane<16 -> N=lane, K=j ; lane>=16 -> N=lane-16,
// K=16+j  (j = element 0..15 of the v16bf).  One thread builds one lane-group.
// ---------------------------------------------------------------------------
__global__ void k_repack(const float* __restrict__ E,
                         __bf16* __restrict__ ebh, __bf16* __restrict__ ebl) {
    int t = blockIdx.x * blockDim.x + threadIdx.x;   // 256*8*32 = 65536 total
    int lane = t & 31;
    int c    = (t >> 5) & 7;      // D-chunk
    int kt   = t >> 8;            // k-tile
    int row   = kt * 16 + (lane & 15);
    int dbase = c * 32 + ((lane >> 4) << 4);
    const float* src = E + (size_t)row * EMB_DIM + dbase;
    __bf16* dh = ebh + (size_t)t * 16;
    __bf16* dl = ebl + (size_t)t * 16;
    #pragma unroll
    for (int j = 0; j < 16; ++j) {
        float v   = src[j];
        __bf16 h  = (__bf16)v;
        dh[j] = h;
        dl[j] = (__bf16)(v - (float)h);
    }
}

// ---------------------------------------------------------------------------
// Kernel 3: fused distance-GEMM + argmin + gather + loss partials.
// One WG = 32 z-rows; 8 waves: wave = (mtile in 0..1) + 2*(ksub in 0..3).
// ---------------------------------------------------------------------------
__launch_bounds__(256)
__global__ void k_main(const float* __restrict__ z,
                       const float* __restrict__ E,
                       const __bf16* __restrict__ ebh,
                       const __bf16* __restrict__ ebl,
                       const float* __restrict__ enorm,
                       float* __restrict__ out,
                       float* __restrict__ partials) {
    __shared__ float zt[EMB_DIM * ROWS];              // [d][row], 32 KB
    __shared__ unsigned long long keys[ROWS];
    __shared__ int   sidx[ROWS];
    __shared__ float wsum[8];

    const int wg = blockIdx.x;
    const int t  = threadIdx.x;
    const int n0  = wg * ROWS;
    const int b   = n0 / HW;
    const int hw0 = n0 % HW;
    const float* zbase = z + (size_t)b * EMB_DIM * HW + hw0;

#if USE_TDM
    // -------- Tensor Data Mover: one DMA stages the whole 2D tile ----------
    // 2D tensor: dim0 (x) = 32 contiguous floats, dim1 (y) = 256 rows with a
    // row stride of HW=1024 floats.  Lands contiguously as zt[d][row].
    if (t < 32) {                       // wave 0 issues the descriptor
        unsigned long long ga = (unsigned long long)(uintptr_t)zbase;
        unsigned int lds_lo   = (unsigned int)(uintptr_t)(&zt[0]);
        v4u g0 = {};
        g0[0] = 1u;                                           // count=1, user
        g0[1] = lds_lo;                                       // lds_addr
        g0[2] = (unsigned int)(ga & 0xFFFFFFFFu);             // gaddr[31:0]
        g0[3] = (unsigned int)((ga >> 32) & 0x01FFFFFFu)      // gaddr[56:32]
              | (2u << 30);                                   // type=2 (image)
        v8i g1 = {};
        g1[0] = (int)(2u << 16);                              // data_size=4B
        g1[1] = (int)((32u & 0xFFFFu) << 16);                 // tensor_dim0 lo16
        g1[2] = (int)((32u >> 16) | ((256u & 0xFFFFu) << 16));// dim0 hi|dim1 lo
        g1[3] = (int)((256u >> 16) | (32u << 16));            // dim1 hi|tile_dim0
        g1[4] = (int)(256u);                                  // tile_dim1=256
        g1[5] = (int)(1024);                                  // dim0_stride lo32
        g1[6] = 0;                                            // stride hi bits
        g1[7] = 0;
        v4i gz4 = {};
        v8i gz8 = {};
        __builtin_amdgcn_tensor_load_to_lds(g0, g1, gz4, gz4, gz8, 0);
        __builtin_amdgcn_s_wait_tensorcnt(0);
    }
#else
    // -------- Fallback: float4-coalesced staging ---------------------------
    #pragma unroll
    for (int i = 0; i < 8; ++i) {
        int f4   = t + i * 256;      // 0..2047
        int col4 = f4 & 7;
        int d    = f4 >> 3;
        float4 v = *(const float4*)(zbase + (size_t)d * HW + col4 * 4);
        *(float4*)(zt + d * ROWS + col4 * 4) = v;
    }
#endif
    if (t < ROWS) keys[t] = ~0ull;
    __syncthreads();

    const int wave  = t >> 5;
    const int lane  = t & 31;
    const int mtile = wave & 1;
    const int ksub  = wave >> 1;          // 0..3, strides the k-tile loop
    const int mrow  = mtile * 16;
    const int arow  = mrow + (lane & 15);
    const int khi   = (lane >> 4) & 1;    // upper half-wave flag

    // Build A-operand register images once (invariant over the K loop).
    // A 16x32 bf16, contiguous k order used here (B matches): lane<16 -> M=lane, K = {0..15};
    //                      lane>=16 -> M=lane-16, K = {16..31}.
    v16bf Ah[CHUNKS], Al[CHUNKS];
    #pragma unroll
    for (int c = 0; c < CHUNKS; ++c) {
        v16bf h, l;
        #pragma unroll
        for (int j = 0; j < 16; ++j) {
            int dloc = j + khi * 16;
            float v  = zt[(c * 32 + dloc) * ROWS + arow];
            __bf16 hh = (__bf16)v;
            h[j] = hh;
            l[j] = (__bf16)(v - (float)hh);
        }
        Ah[c] = h; Al[c] = l;
    }

    // B-operand byte-group address helper (matches k_repack's t*16 layout).
    auto boff = [&](int kt, int c) -> size_t {
        return ((size_t)(kt * CHUNKS + c) * 32 + (size_t)lane) * 16;
    };

    // Software-pipelined codebook sweep: prefetch next chunk's B while the
    // current chunk's three WMMAs execute.
    v16bf Bh = *(const v16bf*)(ebh + boff(ksub, 0));
    v16bf Bl = *(const v16bf*)(ebl + boff(ksub, 0));
    for (int kt = ksub; kt < KTILES; kt += 4) {
        v8f acc = {};
        #pragma unroll
        for (int c = 0; c < CHUNKS; ++c) {
            int nkt = (c == CHUNKS - 1) ? (kt + 4 < KTILES ? kt + 4 : ksub) : kt;
            int nc  = (c == CHUNKS - 1) ? 0 : c + 1;
            v16bf nBh = *(const v16bf*)(ebh + boff(nkt, nc));
            v16bf nBl = *(const v16bf*)(ebl + boff(nkt, nc));
            // z.e  ~=  zh.eh + zl.eh + zh.el   (f32 accumulate)
            acc = __builtin_amdgcn_wmma_f32_16x16x32_bf16(false, Ah[c], false, Bh,
                                                          (short)0, acc, false, false);
            acc = __builtin_amdgcn_wmma_f32_16x16x32_bf16(false, Al[c], false, Bh,
                                                          (short)0, acc, false, false);
            acc = __builtin_amdgcn_wmma_f32_16x16x32_bf16(false, Ah[c], false, Bl,
                                                          (short)0, acc, false, false);
            Bh = nBh; Bl = nBl;
        }
        int   k  = kt * 16 + (lane & 15);
        float en = enorm[k];
        #pragma unroll
        for (int r = 0; r < 8; ++r) {
            // C/D layout: lane -> column (k), VGPR r -> row (r or r+8).
            float dist = en - 2.0f * acc[r];
            unsigned u = __float_as_uint(dist);
            u = (u & 0x80000000u) ? ~u : (u | 0x80000000u);   // orderable bits
            unsigned long long key =
                ((unsigned long long)u << 32) | (unsigned)k;  // ties -> min k
            int rowl = mrow + (khi ? r + 8 : r);
            atomicMin(&keys[rowl], key);                      // ds_min_u64
        }
    }
    __syncthreads();

    if (t < ROWS) {
        int idx = (int)(keys[t] & 0xFFFFFFFFull);
        sidx[t] = idx;
        out[(size_t)ZQ_ELEMS + 1 + n0 + t] = (float)idx;      // indices output
    }
    __syncthreads();

    // Gather z_q (== straight-through output), write NCHW, loss partial.
    float lsum = 0.f;
    #pragma unroll 4
    for (int i = 0; i < 32; ++i) {
        int flat = t + i * 256;        // 32 rows * 256 d = 8192 elements
        int row  = flat & 31;
        int d    = flat >> 5;
        float zq  = E[(size_t)sidx[row] * EMB_DIM + d];
        float zp  = zt[d * ROWS + row];
        float dif = zq - zp;
        lsum += dif * dif;
        out[(size_t)b * EMB_DIM * HW + (size_t)d * HW + hw0 + row] = zq;
    }
    for (int off = 16; off; off >>= 1) lsum += __shfl_down(lsum, off);
    if (lane == 0) wsum[wave] = lsum;
    __syncthreads();
    if (t == 0) {
        float s = 0.f;
        #pragma unroll
        for (int i = 0; i < 8; ++i) s += wsum[i];
        partials[wg] = s;
    }
}

// ---------------------------------------------------------------------------
// Kernel 4: deterministic loss reduction.  loss = (1+BETA) * mean(diff^2)
// ---------------------------------------------------------------------------
__global__ void k_loss(const float* __restrict__ partials, float* __restrict__ out,
                       int nparts) {
    __shared__ float ws[8];
    int t = threadIdx.x;  // 256
    float s = 0.f;
    for (int i = t; i < nparts; i += 256) s += partials[i];
    for (int off = 16; off; off >>= 1) s += __shfl_down(s, off);
    if ((t & 31) == 0) ws[t >> 5] = s;
    __syncthreads();
    if (t == 0) {
        float tot = 0.f;
        #pragma unroll
        for (int i = 0; i < 8; ++i) tot += ws[i];
        out[ZQ_ELEMS] = tot * 11.0f / (float)ZQ_ELEMS;   // (1 + BETA) * mean
    }
}

// ---------------------------------------------------------------------------
extern "C" void kernel_launch(void* const* d_in, const int* in_sizes, int n_in,
                              void* d_out, int out_size, void* d_ws, size_t ws_size,
                              hipStream_t stream) {
    const float* z = (const float*)d_in[0];        // [32,256,32,32]
    const float* E = (const float*)d_in[1];        // [4096,256]
    float* out = (float*)d_out;                    // z_q | loss | indices

    // Workspace layout (needs ~4.2 MB):
    //   [0, 2MB)        ebh : bf16 hi image of E in B-operand layout
    //   [2MB, 4MB)      ebl : bf16 lo image
    //   [4MB, 4MB+16KB) enorm : 4096 f32
    //   then            partials : 1024 f32
    char* ws = (char*)d_ws;
    __bf16* ebh   = (__bf16*)ws;
    __bf16* ebl   = (__bf16*)(ws + (size_t)N_EMB * EMB_DIM * 2);
    float*  enorm = (float*)(ws + (size_t)N_EMB * EMB_DIM * 4);
    float*  parts = enorm + N_EMB;

    k_enorm <<<N_EMB, 32, 0, stream>>>(E, enorm);
    k_repack<<<(KTILES * CHUNKS * 32) / 256, 256, 0, stream>>>(E, ebh, ebl);
    k_main  <<<NVEC / ROWS, 256, 0, stream>>>(z, E, ebh, ebl, enorm, out, parts);
    k_loss  <<<1, 256, 0, stream>>>(parts, out, NVEC / ROWS);
}
